// ParticleGNNAdvanced_25812753449158
// MI455X (gfx1250) — matrix operand order repaired and checked by reading, not compiled, not run on hardware
//
#include <hip/hip_runtime.h>
#include <hip/hip_bf16.h>
#include <math.h>

#define HEADS 4

typedef __attribute__((ext_vector_type(16))) _Float16 v16h;
typedef __attribute__((ext_vector_type(8)))  _Float16 v8h;
typedef __attribute__((ext_vector_type(8)))  float    v8f;

__device__ __forceinline__ unsigned f2ord(float f) {
  unsigned u = __float_as_uint(f);
  return (u & 0x80000000u) ? ~u : (u | 0x80000000u);
}
__device__ __forceinline__ float ord2f(unsigned u) {
  return (u & 0x80000000u) ? __uint_as_float(u & 0x7FFFFFFFu)
                           : __uint_as_float(~u);
}
__device__ __forceinline__ float lrelu(float v) { return v > 0.f ? v : 0.2f * v; }

// ---------------------------------------------------------------- fill
__global__ void k_fill(float* p, float v, int n) {
  int i = blockIdx.x * blockDim.x + threadIdx.x;
  if (i < n) p[i] = v;
}

// ---------------------------------------------------------------- WMMA GEMM
// C[M,Nc] = A[M,K] @ B[K,Nc] + bias. f32 in/out, f16 WMMA, f32 accumulate.
// 128 threads = 4 waves; block tile 64x16 (wave w owns rows w*16..w*16+15).
// K and Nc are compile-time (Nc % 16 == 0, grid.y = Nc/16, so no column
// guards at all); k-chunk loop fully unrolls into back-to-back v_wmma.
// Bias pointer is always valid (zeros vector used for bias-less GEMMs).
template <int K, int Nc>
__global__ __launch_bounds__(128)
void k_gemm_wmma(const float* __restrict__ A, const float* __restrict__ B,
                 const float* __restrict__ bias, float* __restrict__ C, int M) {
  constexpr int KP = (K + 31) & ~31;                // K padded to 32
  __shared__ __align__(16) _Float16 Asm[64 * KP];   // row-major, stride KP
  __shared__ __align__(16) _Float16 Bsm[16 * KP];   // B transposed: [col][k]

  const int t      = threadIdx.x;
  const int wave   = t >> 5;
  const int lane   = t & 31;
  const int half16 = lane >> 4;
  const int l16    = lane & 15;
  const int mBase  = blockIdx.x * 64;
  const int nBase  = blockIdx.y * 16;

  // ---- stage A tile (64 x KP): unconditional clamped loads + cndmask zeroing ----
  #pragma unroll 4
  for (int idx = t; idx < 64 * KP; idx += 128) {
    int r = idx / KP, k = idx % KP;
    int gr = mBase + r;
    int cr = gr < M ? gr : (M - 1);
    int ck = (KP == K) ? k : (k < K ? k : (K - 1));
    float v = A[(size_t)cr * K + ck];
    v = (gr < M && k < K) ? v : 0.f;
    Asm[idx] = (_Float16)v;
  }
  // ---- stage B tile transposed (16 cols x KP); col always in range ----
  #pragma unroll 4
  for (int idx = t; idx < 16 * KP; idx += 128) {
    int c = idx / KP, k = idx % KP;
    int ck = (KP == K) ? k : (k < K ? k : (K - 1));
    float v = B[(size_t)ck * Nc + nBase + c];
    v = (k < K) ? v : 0.f;
    Bsm[idx] = (_Float16)v;
  }
  __syncthreads();

  const _Float16* arow = &Asm[(size_t)(wave * 16 + l16) * KP];
  const _Float16* bcol = &Bsm[(size_t)l16 * KP];

  v8f acc = {0.f, 0.f, 0.f, 0.f, 0.f, 0.f, 0.f, 0.f};

  #pragma unroll
  for (int k0 = 0; k0 < KP; k0 += 32) {
    // A fragment: lanes 0-15 hold K {0..7,16..23}, lanes 16-31 hold {8..15,24..31}
    v8h a0 = *(const v8h*)&arow[k0 + half16 * 8];
    v8h a1 = *(const v8h*)&arow[k0 + 16 + half16 * 8];
    // B fragment: lanes 0-15 hold K {0..7,16..23}, lanes 16-31 hold {8..15,24..31}
    v8h b0 = *(const v8h*)&bcol[k0 + half16 * 8];
    v8h b1 = *(const v8h*)&bcol[k0 + 16 + half16 * 8];
    v16h af = __builtin_shufflevector(a0, a1, 0, 1, 2, 3, 4, 5, 6, 7,
                                      8, 9, 10, 11, 12, 13, 14, 15);
    v16h bf = __builtin_shufflevector(b0, b1, 0, 1, 2, 3, 4, 5, 6, 7,
                                      8, 9, 10, 11, 12, 13, 14, 15);
    acc = __builtin_amdgcn_wmma_f32_16x16x32_f16(
        /*neg_a=*/false, af, /*neg_b=*/false, bf,
        /*c_mod=*/(short)0, acc, /*reuse_a=*/false, /*reuse_b=*/false);
  }

  const int col = nBase + l16;                 // always < Nc
  const float bc = bias[col];                  // unconditional
  const int r0 = mBase + wave * 16 + half16 * 8;
  float* crow = C + (size_t)r0 * Nc + col;
  if (mBase + wave * 16 + 16 <= M) {           // wave-uniform fast path
    #pragma unroll
    for (int j = 0; j < 8; j++) crow[(size_t)j * Nc] = acc[j] + bc;
  } else {
    #pragma unroll
    for (int j = 0; j < 8; j++)
      if (r0 + j < M) crow[(size_t)j * Nc] = acc[j] + bc;
  }
}

// ---------------------------------------------------------------- edge encoder
__global__ void k_edge_enc(const float* __restrict__ ea,
                           const float* __restrict__ W1, const float* __restrict__ b1,
                           const float* __restrict__ W2, const float* __restrict__ b2,
                           const float* __restrict__ W3, const float* __restrict__ b3,
                           const int* __restrict__ dst,
                           float* __restrict__ ef, float* __restrict__ loopsum,
                           float* __restrict__ cnt, int E) {
  int e = blockIdx.x * blockDim.x + threadIdx.x;
  if (e >= E) return;
  float a = ea[e];
  float h1[16], h2[16];
  #pragma unroll
  for (int k = 0; k < 16; k++) h1[k] = fmaxf(a * W1[k] + b1[k], 0.f);
  #pragma unroll
  for (int k = 0; k < 16; k++) {
    float s = b2[k];
    #pragma unroll
    for (int j = 0; j < 16; j++) s += h1[j] * W2[j * 16 + k];
    h2[k] = fmaxf(s, 0.f);
  }
  int d = dst[e];
  #pragma unroll
  for (int k = 0; k < 4; k++) {
    float s = b3[k];
    #pragma unroll
    for (int j = 0; j < 16; j++) s += h2[j] * W3[j * 4 + k];
    ef[(size_t)e * 4 + k] = s;
    atomicAdd(&loopsum[(size_t)d * 4 + k], s);
  }
  atomicAdd(&cnt[d], 1.f);
}

__global__ void k_loopdiv(float* loopsum, const float* cnt, int N) {
  int n = blockIdx.x * blockDim.x + threadIdx.x;
  if (n >= N) return;
  float c = fmaxf(cnt[n], 1.f);
  #pragma unroll
  for (int k = 0; k < 4; k++) loopsum[(size_t)n * 4 + k] /= c;
}

// ---------------------------------------------------------------- per-node attention dot
__global__ void k_att(const float* __restrict__ xt, const float* __restrict__ att,
                      float* __restrict__ out, int N, int C) {
  int i = blockIdx.x * blockDim.x + threadIdx.x;
  if (i >= N * HEADS) return;
  int n = i >> 2, h = i & 3;
  const float* r = xt + (size_t)n * (HEADS * C) + h * C;
  const float* a = att + h * C;
  float s = 0.f;
  for (int c = 0; c < C; c++) s += r[c] * a[c];
  out[i] = s;
}

// M1[j][h] = sum_c lin_edge[j, h*C+c] * att_edge[h, c]
__global__ void k_m1(const float* __restrict__ linE, const float* __restrict__ attE,
                     float* __restrict__ M1, int C) {
  int t = threadIdx.x;
  if (t >= 16) return;
  int j = t >> 2, h = t & 3;
  const float* row = linE + (size_t)j * (HEADS * C) + h * C;
  const float* a = attE + h * C;
  float s = 0.f;
  for (int c = 0; c < C; c++) s += row[c] * a[c];
  M1[j * 4 + h] = s;
}

// ---------------------------------------------------------------- edge pass 1: logits + seg-max
__global__ void k_pass1(const int* __restrict__ src, const int* __restrict__ dst,
                        const float* __restrict__ ef, const float* __restrict__ loopattr,
                        const float* __restrict__ ts, const float* __restrict__ td,
                        const float* __restrict__ M1, float* __restrict__ logit,
                        unsigned* __restrict__ mord, int E, int Nn, int useLoop) {
  int e = blockIdx.x * blockDim.x + threadIdx.x;
  if (e >= E + Nn) return;
  int s, d;
  float e0, e1, e2, e3;
  if (e < E) {
    s = src[e]; d = dst[e];
    e0 = ef[(size_t)e * 4 + 0]; e1 = ef[(size_t)e * 4 + 1];
    e2 = ef[(size_t)e * 4 + 2]; e3 = ef[(size_t)e * 4 + 3];
  } else {
    s = d = e - E;
    if (useLoop) {
      e0 = loopattr[(size_t)s * 4 + 0]; e1 = loopattr[(size_t)s * 4 + 1];
      e2 = loopattr[(size_t)s * 4 + 2]; e3 = loopattr[(size_t)s * 4 + 3];
    } else { e0 = e1 = e2 = e3 = 0.f; }
  }
  #pragma unroll
  for (int h = 0; h < 4; h++) {
    float ae = e0 * M1[h] + e1 * M1[4 + h] + e2 * M1[8 + h] + e3 * M1[12 + h];
    float lg = lrelu(ts[(size_t)s * 4 + h] + td[(size_t)d * 4 + h] + ae);
    logit[(size_t)e * 4 + h] = lg;
    atomicMax(&mord[(size_t)d * 4 + h], f2ord(lg));
  }
}

// ---------------------------------------------------------------- edge pass 2 (GAT)
__global__ void k_pass2_gat(const int* __restrict__ src, const int* __restrict__ dst,
                            const float* __restrict__ logit, const unsigned* __restrict__ mord,
                            float* __restrict__ ssum, const float* __restrict__ xt,
                            float* __restrict__ outun, int E, int Nn, int C) {
  int t = blockIdx.x * blockDim.x + threadIdx.x;
  if (t >= (E + Nn) * 4) return;
  int e = t >> 2, h = t & 3;
  int s, d;
  if (e < E) { s = src[e]; d = dst[e]; } else { s = d = e - E; }
  float w = __expf(logit[(size_t)e * 4 + h] - ord2f(mord[(size_t)d * 4 + h]));
  atomicAdd(&ssum[(size_t)d * 4 + h], w);
  const float* xr = xt + (size_t)s * (4 * C) + h * C;
  float* orow = outun + (size_t)d * (4 * C) + h * C;
  __builtin_prefetch(xr, 0, 0);
  for (int c = 0; c < C; c++) atomicAdd(&orow[c], w * xr[c]);
}

// edge pass 2 (EdgeGAT, C==48): message = alpha*(xs[src] + ea @ lin_edge)
__global__ void k_pass2_egat(const int* __restrict__ src, const int* __restrict__ dst,
                             const float* __restrict__ logit, const unsigned* __restrict__ mord,
                             float* __restrict__ ssum, const float* __restrict__ xs,
                             const float* __restrict__ ef, const float* __restrict__ linE,
                             float* __restrict__ outun, int E, int Nn) {
  const int C = 48;
  int t = blockIdx.x * blockDim.x + threadIdx.x;
  if (t >= (E + Nn) * 4) return;
  int e = t >> 2, h = t & 3;
  int s, d;
  float e0 = 0.f, e1 = 0.f, e2 = 0.f, e3 = 0.f;
  bool real = (e < E);
  if (real) {
    s = src[e]; d = dst[e];
    e0 = ef[(size_t)e * 4 + 0]; e1 = ef[(size_t)e * 4 + 1];
    e2 = ef[(size_t)e * 4 + 2]; e3 = ef[(size_t)e * 4 + 3];
  } else { s = d = e - E; }
  float w = __expf(logit[(size_t)e * 4 + h] - ord2f(mord[(size_t)d * 4 + h]));
  atomicAdd(&ssum[(size_t)d * 4 + h], w);
  const float* xr = xs + (size_t)s * (4 * C) + h * C;
  const float* L0 = linE + 0 * (4 * C) + h * C;
  const float* L1 = linE + 1 * (4 * C) + h * C;
  const float* L2 = linE + 2 * (4 * C) + h * C;
  const float* L3 = linE + 3 * (4 * C) + h * C;
  float* orow = outun + (size_t)d * (4 * C) + h * C;
  __builtin_prefetch(xr, 0, 0);
  for (int c = 0; c < C; c++) {
    float msg = xr[c];
    if (real) msg += e0 * L0[c] + e1 * L1[c] + e2 * L2[c] + e3 * L3[c];
    atomicAdd(&orow[c], w * msg);
  }
}

// ---------------------------------------------------------------- finalize convs
__global__ void k_finish_gat(const float* __restrict__ outun, const float* __restrict__ ssum,
                             const float* __restrict__ bias, float* __restrict__ raw,
                             int N, int C) {
  int i = blockIdx.x * blockDim.x + threadIdx.x;
  if (i >= N * C) return;
  int n = i / C, c = i % C;
  float acc = 0.f;
  #pragma unroll
  for (int h = 0; h < 4; h++)
    acc += outun[(size_t)n * 4 * C + h * C + c] / (ssum[(size_t)n * 4 + h] + 1e-16f);
  raw[i] = 0.25f * acc + bias[c];
}

__global__ void k_finish_egat(const float* __restrict__ outun, const float* __restrict__ ssum,
                              const float* __restrict__ bias, float* __restrict__ raw, int N) {
  int i = blockIdx.x * blockDim.x + threadIdx.x;
  if (i >= N * 192) return;
  int n = i / 192, k = i % 192;
  int h = k / 48;
  raw[i] = outun[i] / (ssum[(size_t)n * 4 + h] + 1e-16f) + bias[k];
}

// ---------------------------------------------------------------- layer norm (+optional relu)
__global__ void k_ln(const float* __restrict__ in, const float* __restrict__ g,
                     const float* __restrict__ b, float* __restrict__ out,
                     int N, int C, int act) {
  int n = blockIdx.x * blockDim.x + threadIdx.x;
  if (n >= N) return;
  const float* r = in + (size_t)n * C;
  float m = 0.f;
  for (int c = 0; c < C; c++) m += r[c];
  m /= (float)C;
  float v = 0.f;
  for (int c = 0; c < C; c++) { float d = r[c] - m; v += d * d; }
  v /= (float)C;
  float rc = rsqrtf(v + 1e-5f);
  float* o = out + (size_t)n * C;
  for (int c = 0; c < C; c++) {
    float y = (r[c] - m) * rc * g[c] + b[c];
    o[c] = act ? fmaxf(y, 0.f) : y;
  }
}

// ---------------------------------------------------------------- gated residual
__global__ void k_gate(const float* __restrict__ xp, const float* __restrict__ idp,
                       const float* __restrict__ W, const float* __restrict__ gb,
                       float* __restrict__ out, int N, int D) {
  int i = blockIdx.x * blockDim.x + threadIdx.x;
  if (i >= N * D) return;
  int n = i / D, co = i % D;
  const float* xr = xp + (size_t)n * D;
  const float* ir = idp + (size_t)n * D;
  float a = gb[co];
  for (int k = 0; k < D; k++)
    a += xr[k] * W[(size_t)k * D + co] + ir[k] * W[(size_t)(D + k) * D + co];
  float gv = 1.f / (1.f + __expf(-a));
  out[i] = gv * xr[co] + (1.f - gv) * ir[co];
}

// ---------------------------------------------------------------- global pooling
__global__ void k_pool(const float* __restrict__ x4, float* __restrict__ gsum,
                       unsigned* __restrict__ gmax, int N, int C) {
  int t = blockIdx.x * blockDim.x + threadIdx.x;
  if (t >= C * 64) return;
  int c = t % C, p = t / C;
  float s = 0.f, mx = -3.402823466e38f;
  bool any = false;
  for (int n = p; n < N; n += 64) {
    float v = x4[(size_t)n * C + c];
    s += v;
    if (v > mx) mx = v;
    any = true;
  }
  atomicAdd(&gsum[c], s);
  if (any) atomicMax(&gmax[c], f2ord(mx));
}

// ---------------------------------------------------------------- classifier head
__global__ void k_classifier(const float* __restrict__ gsum, const unsigned* __restrict__ gmax,
                             const float* W1, const float* b1, const float* g1, const float* bb1,
                             const float* W2, const float* b2, const float* g2, const float* bb2,
                             const float* W3, const float* b3, float* __restrict__ cls, int N) {
  __shared__ float pIn[288], h1[96], h2[48], o3[3];
  int t = threadIdx.x;  // blockDim.x == 96
  for (int i = t; i < 288; i += 96) {
    float v;
    if (i < 96) v = gsum[i] / (float)N;
    else if (i < 192) v = ord2f(gmax[i - 96]);
    else v = gsum[i - 192];
    pIn[i] = v;
  }
  __syncthreads();
  float a1 = b1[t];
  for (int k = 0; k < 288; k++) a1 += pIn[k] * W1[k * 96 + t];
  h1[t] = a1;
  __syncthreads();
  float m = 0.f;
  for (int k = 0; k < 96; k++) m += h1[k];
  m /= 96.f;
  float v = 0.f;
  for (int k = 0; k < 96; k++) { float d = h1[k] - m; v += d * d; }
  v /= 96.f;
  float r = rsqrtf(v + 1e-5f);
  float hv = fmaxf((h1[t] - m) * r * g1[t] + bb1[t], 0.f);
  __syncthreads();
  h1[t] = hv;
  __syncthreads();
  float a2 = 0.f;
  if (t < 48) {
    a2 = b2[t];
    for (int k = 0; k < 96; k++) a2 += h1[k] * W2[k * 48 + t];
  }
  __syncthreads();
  if (t < 48) h2[t] = a2;
  __syncthreads();
  float val = 0.f;
  if (t < 48) {
    float m2 = 0.f;
    for (int k = 0; k < 48; k++) m2 += h2[k];
    m2 /= 48.f;
    float v2 = 0.f;
    for (int k = 0; k < 48; k++) { float d = h2[k] - m2; v2 += d * d; }
    v2 /= 48.f;
    float r2 = rsqrtf(v2 + 1e-5f);
    val = fmaxf((h2[t] - m2) * r2 * g2[t] + bb2[t], 0.f);
  }
  __syncthreads();
  if (t < 48) h2[t] = val;
  __syncthreads();
  if (t < 3) {
    float a3 = b3[t];
    for (int k = 0; k < 48; k++) a3 += h2[k] * W3[k * 3 + t];
    o3[t] = a3;
  }
  __syncthreads();
  if (t < 3) {
    float mm = fmaxf(fmaxf(o3[0], o3[1]), o3[2]);
    float se = __expf(o3[0] - mm) + __expf(o3[1] - mm) + __expf(o3[2] - mm);
    cls[t] = o3[t] - mm - logf(se);
  }
}

__global__ void k_broadcast(const float* __restrict__ cls, float* __restrict__ out, int N) {
  int i = blockIdx.x * blockDim.x + threadIdx.x;
  if (i < N * 3) out[i] = cls[i % 3];
}

// ================================================================ host
extern "C" void kernel_launch(void* const* d_in, const int* in_sizes, int n_in,
                              void* d_out, int out_size, void* d_ws, size_t ws_size,
                              hipStream_t stream) {
  const float* x     = (const float*)d_in[0];
  const int*   ei    = (const int*)d_in[1];
  const float* eattr = (const float*)d_in[2];

  const int N  = in_sizes[0] / 8;
  const int E  = in_sizes[2];       // edge_attr is [E,1]
  const int E2 = E + N;
  const int* src = ei;
  const int* dst = ei + E;

  // ---- params: jax pytree leaf order (dict keys sorted; lists/tuples in order) ----
  const float* P[62];
  for (int i = 0; i < 62 && (i + 3) < n_in; i++) P[i] = (const float*)d_in[3 + i];
  const float *clW1 = P[0], *clb1 = P[1], *clg1 = P[2], *clbb1 = P[3];
  const float *clW2 = P[4], *clb2 = P[5], *clg2 = P[6], *clbb2 = P[7];
  const float *clW3 = P[8], *clb3 = P[9];
  const float *c1aD = P[10], *c1aE = P[11], *c1aS = P[12], *c1b = P[13], *c1lin = P[14], *c1lE = P[15];
  const float *c2aD = P[16], *c2aE = P[17], *c2aS = P[18], *c2b = P[19], *c2lD = P[20], *c2lE = P[21], *c2lS = P[22];
  const float *c3aD = P[23], *c3aE = P[24], *c3aS = P[25], *c3b = P[26], *c3lin = P[27], *c3lE = P[28];
  const float *c4aD = P[29], *c4aE = P[30], *c4aS = P[31], *c4b = P[32], *c4lD = P[33], *c4lE = P[34], *c4lS = P[35];
  const float *eeW1 = P[36], *eeb1 = P[37], *eeW2 = P[38], *eeb2 = P[39], *eeW3 = P[40], *eeb3 = P[41];
  const float *g1W = P[42], *g1b = P[43], *g2W = P[44], *g2b = P[45];
  const float *inG = P[46], *inB = P[47];
  const float *ipW = P[48], *ipB = P[49];
  const float *n1G = P[50], *n1B = P[51], *n2G = P[52], *n2B = P[53];
  const float *n3G = P[54], *n3B = P[55], *n4G = P[56], *n4B = P[57];
  const float *p2W = P[58], *p2B = P[59], *p4W = P[60], *p4B = P[61];

  // ---- workspace layout ----
  float* ws = (float*)d_ws;
  size_t off = 0;
  auto alloc = [&](size_t n) { float* p = ws + off; off += (n + 3) & ~(size_t)3; return p; };
  float*    ef       = alloc((size_t)E * 4);
  float*    loopattr = alloc((size_t)N * 4);
  float*    cnt      = alloc((size_t)N);
  float*    tS       = alloc((size_t)N * 4);
  float*    tD       = alloc((size_t)N * 4);
  unsigned* mord     = (unsigned*)alloc((size_t)N * 4);
  float*    ssum     = alloc((size_t)N * 4);
  float*    logit    = alloc((size_t)E2 * 4);
  float*    T1       = alloc((size_t)N * 384); // xt / xs / raw / LN buffer
  float*    T2       = alloc((size_t)N * 384); // xd -> out_un accumulator
  float*    hbuf     = alloc((size_t)N * 48);
  float*    x1       = alloc((size_t)N * 48);
  float*    x2       = alloc((size_t)N * 48);
  float*    x3       = alloc((size_t)N * 96);
  float*    x4       = alloc((size_t)N * 96);
  float*    tmpn     = alloc((size_t)N * 96);
  float*    M1b      = alloc(16);
  float*    gsum     = alloc(96);
  unsigned* gmax     = (unsigned*)alloc(96);
  float*    cls      = alloc(4);
  float*    zbias    = alloc(384);             // zero bias for bias-less GEMMs
  (void)ws_size; (void)out_size;

  auto cdiv = [](long a, long b) { return (int)((a + b - 1) / b); };
  auto fill = [&](float* p, float v, long n) {
    k_fill<<<dim3(cdiv(n, 256)), dim3(256), 0, stream>>>(p, v, (int)n);
  };
  auto gemm = [&](const float* A, const float* B, const float* bias, float* C,
                  int M, int K, int Nc) {
    const float* bp = bias ? bias : zbias;
    dim3 g(cdiv(M, 64), Nc / 16), b(128);
    if      (K == 8   && Nc == 48)  k_gemm_wmma<8, 48><<<g, b, 0, stream>>>(A, B, bp, C, M);
    else if (K == 48  && Nc == 192) k_gemm_wmma<48, 192><<<g, b, 0, stream>>>(A, B, bp, C, M);
    else if (K == 48  && Nc == 384) k_gemm_wmma<48, 384><<<g, b, 0, stream>>>(A, B, bp, C, M);
    else if (K == 96  && Nc == 192) k_gemm_wmma<96, 192><<<g, b, 0, stream>>>(A, B, bp, C, M);
    else if (K == 192 && Nc == 48)  k_gemm_wmma<192, 48><<<g, b, 0, stream>>>(A, B, bp, C, M);
    else if (K == 192 && Nc == 96)  k_gemm_wmma<192, 96><<<g, b, 0, stream>>>(A, B, bp, C, M);
  };

  fill(zbias, 0.f, 384);

  // ---- edge encoder + loop_attr ----
  fill(loopattr, 0.f, (long)N * 4);
  fill(cnt, 0.f, N);
  k_edge_enc<<<dim3(cdiv(E, 256)), dim3(256), 0, stream>>>(
      eattr, eeW1, eeb1, eeW2, eeb2, eeW3, eeb3, dst, ef, loopattr, cnt, E);
  k_loopdiv<<<dim3(cdiv(N, 256)), dim3(256), 0, stream>>>(loopattr, cnt, N);

  // ---- input projection + LN + relu ----
  gemm(x, ipW, ipB, T1, N, 8, 48);
  k_ln<<<dim3(cdiv(N, 256)), dim3(256), 0, stream>>>(T1, inG, inB, hbuf, N, 48, 1);

  // ---- GATConv ----
  auto runGat = [&](const float* xin, int Cin, int C,
                    const float* lin, const float* attS, const float* attD,
                    const float* attE, const float* linE, const float* bias,
                    const float* ng, const float* nb, float* xout) {
    int HC = 4 * C;
    gemm(xin, lin, nullptr, T1, N, Cin, HC);
    k_att<<<dim3(cdiv((long)N * 4, 256)), dim3(256), 0, stream>>>(T1, attS, tS, N, C);
    k_att<<<dim3(cdiv((long)N * 4, 256)), dim3(256), 0, stream>>>(T1, attD, tD, N, C);
    k_m1<<<dim3(1), dim3(16), 0, stream>>>(linE, attE, M1b, C);
    fill((float*)mord, 0.f, (long)N * 4);
    fill(ssum, 0.f, (long)N * 4);
    fill(T2, 0.f, (long)N * HC);
    k_pass1<<<dim3(cdiv(E2, 256)), dim3(256), 0, stream>>>(
        src, dst, ef, loopattr, tS, tD, M1b, logit, mord, E, N, /*useLoop=*/1);
    k_pass2_gat<<<dim3(cdiv((long)E2 * 4, 256)), dim3(256), 0, stream>>>(
        src, dst, logit, mord, ssum, T1, T2, E, N, C);
    k_finish_gat<<<dim3(cdiv((long)N * C, 256)), dim3(256), 0, stream>>>(
        T2, ssum, bias, T1, N, C);
    k_ln<<<dim3(cdiv(N, 256)), dim3(256), 0, stream>>>(T1, ng, nb, xout, N, C, 1);
  };

  // ---- EdgeGATConv + LN + proj + gate ----
  auto runEgat = [&](const float* xin, int Cin,
                     const float* linS, const float* linD, const float* linE,
                     const float* attS, const float* attD, const float* attE,
                     const float* bias, const float* ng, const float* nb,
                     const float* pw, const float* pb,
                     const float* gw, const float* gb, const float* identity,
                     int D, float* xout) {
    gemm(xin, linS, nullptr, T1, N, Cin, 192);   // xs
    gemm(xin, linD, nullptr, T2, N, Cin, 192);   // xd
    k_att<<<dim3(cdiv((long)N * 4, 256)), dim3(256), 0, stream>>>(T2, attS, tD, N, 48);
    k_att<<<dim3(cdiv((long)N * 4, 256)), dim3(256), 0, stream>>>(T1, attD, tS, N, 48);
    k_m1<<<dim3(1), dim3(16), 0, stream>>>(linE, attE, M1b, 48);
    fill((float*)mord, 0.f, (long)N * 4);
    fill(ssum, 0.f, (long)N * 4);
    fill(T2, 0.f, (long)N * 192);                // xd dead; reuse as out_un
    k_pass1<<<dim3(cdiv(E2, 256)), dim3(256), 0, stream>>>(
        src, dst, ef, loopattr, tS, tD, M1b, logit, mord, E, N, /*useLoop=*/0);
    k_pass2_egat<<<dim3(cdiv((long)E2 * 4, 256)), dim3(256), 0, stream>>>(
        src, dst, logit, mord, ssum, T1, ef, linE, T2, E, N);
    k_finish_egat<<<dim3(cdiv((long)N * 192, 256)), dim3(256), 0, stream>>>(
        T2, ssum, bias, T1, N);
    k_ln<<<dim3(cdiv(N, 256)), dim3(256), 0, stream>>>(T1, ng, nb, T1, N, 192, 1);
    gemm(T1, pw, pb, tmpn, N, 192, D);
    k_gate<<<dim3(cdiv((long)N * D, 256)), dim3(256), 0, stream>>>(
        tmpn, identity, gw, gb, xout, N, D);
  };

  runGat(hbuf, 48, 48, c1lin, c1aS, c1aD, c1aE, c1lE, c1b, n1G, n1B, x1);
  runEgat(x1, 48, c2lS, c2lD, c2lE, c2aS, c2aD, c2aE, c2b, n2G, n2B,
          p2W, p2B, g1W, g1b, hbuf, 48, x2);
  runGat(x2, 48, 96, c3lin, c3aS, c3aD, c3aE, c3lE, c3b, n3G, n3B, x3);
  runEgat(x3, 96, c4lS, c4lD, c4lE, c4aS, c4aD, c4aE, c4b, n4G, n4B,
          p4W, p4B, g2W, g2b, x3, 96, x4);

  fill(gsum, 0.f, 96);
  fill((float*)gmax, 0.f, 96);
  k_pool<<<dim3(cdiv(96 * 64, 256)), dim3(256), 0, stream>>>(x4, gsum, gmax, N, 96);
  k_classifier<<<dim3(1), dim3(96), 0, stream>>>(
      gsum, gmax, clW1, clb1, clg1, clbb1, clW2, clb2, clg2, clbb2, clW3, clb3, cls, N);
  k_broadcast<<<dim3(cdiv((long)N * 3, 256)), dim3(256), 0, stream>>>(cls, (float*)d_out, N);
}
